// EdgeGNN_8735963480245
// MI455X (gfx1250) — matrix operand order repaired and checked by reading, not compiled, not run on hardware
//
#include <hip/hip_runtime.h>
#include <hip/hip_bf16.h>

typedef __attribute__((ext_vector_type(16))) __bf16 v16bf;
typedef __attribute__((ext_vector_type(8)))  __bf16 v8bf;
typedef __attribute__((ext_vector_type(8)))  float  v8f;

#define DEVINL __device__ __forceinline__

constexpr int B_   = 32;
constexpr int N_   = 100;
constexpr int E_   = 4950;
constexpr int HN_  = 256;
constexpr int HE_  = 128;
constexpr int CIN_ = 64;
constexpr int COUT_= 64;
constexpr int MNODE = B_ * N_;   // 3200  (divisible by 64)
constexpr int MEDGE = B_ * E_;   // 158400 (divisible by 64)

// ---------------------------------------------------------------- GELU (tanh approx, jax.nn.gelu default)
DEVINL float gelu_f(float x) {
    float x3 = x * x * x;
    return 0.5f * x * (1.0f + tanhf(0.7978845608028654f * (x + 0.044715f * x3)));
}

// ---------------------------------------------------------------- WMMA fragment loads (contiguous k order, k = 16*(lane/16) + s, same for A and B)
// A (16x32 bf16, row-major source, stride ld elems):
//   lanes 0-15: row=l, K = k0..k0+15 contiguous
//   lanes 16-31: row=l, K = k0+16..k0+31 contiguous
DEVINL v16bf load_a_frag(const __bf16* src, int row0, int k0, int ld, int lane) {
    int l = lane & 15, hi = lane >> 4;
    const __bf16* p = src + (size_t)(row0 + l) * ld + k0 + hi * 16;
    v8bf c0 = *(const v8bf*)(p);
    v8bf c1 = *(const v8bf*)(p + 8);
    v16bf f;
#pragma unroll
    for (int i = 0; i < 8; i++) { f[i] = c0[i]; f[i + 8] = c1[i]; }
    return f;
}

// B (32x16 bf16) from transposed weights Wt[N][K] (so a "column" is a contiguous row of Wt):
//   lanes 0-15: col=l, K = k0..k0+15 contiguous; lanes 16-31: col=l, K = k0+16..k0+31
DEVINL v16bf load_b_frag(const __bf16* wt, int col0, int k0, int ld, int lane) {
    int l = lane & 15, hi = lane >> 4;
    return *(const v16bf*)(wt + (size_t)(col0 + l) * ld + k0 + hi * 16);
}

// ---------------------------------------------------------------- weight prep: fp32 W[K][N] -> bf16 Wt[N][K]
__global__ __launch_bounds__(256) void wprep_kernel(const float* __restrict__ w,
                                                    __bf16* __restrict__ wt,
                                                    int K, int Nc) {
    int idx = blockIdx.x * 256 + threadIdx.x;
    if (idx >= K * Nc) return;
    int k = idx / Nc, n = idx % Nc;
    wt[(size_t)n * K + k] = (__bf16)w[idx];
}

// ---------------------------------------------------------------- masked mean of incident edge features
// agg[b][n][c] = sum_{m!=n} mask[e(n,m)] * ef[e(n,m)][c] / max(sum mask, 1)
// triu(k=1) linear index: e(i<j) = i*N - i(i+1)/2 + (j-i-1)
__global__ __launch_bounds__(128) void agg_kernel(const float* __restrict__ ef,
                                                  const float* __restrict__ mask,
                                                  __bf16* __restrict__ agg_bf) {
    int bn = blockIdx.x;
    int b = bn / N_, n = bn % N_;
    int c = threadIdx.x;  // 0..127
    const float* efb = ef + (size_t)b * E_ * HE_;
    const float* mb  = mask + (size_t)b * E_;
    float s = 0.f, cnt = 0.f;
    for (int m = 0; m < N_; m++) {
        if (m == n) continue;
        int i = (n < m) ? n : m;
        int j = (n < m) ? m : n;
        int e = i * N_ - (i * (i + 1)) / 2 + (j - i - 1);
        float mv = mb[e];
        cnt += mv;
        s   += mv * efb[(size_t)e * HE_ + c];
    }
    agg_bf[(size_t)bn * HE_ + c] = (__bf16)(s / fmaxf(cnt, 1.0f));
}

// ---------------------------------------------------------------- LayerNorm (fp32 in, bf16*g+b out)
template <int C>
__global__ __launch_bounds__(C) void layernorm_kernel(const float* __restrict__ x,
                                                      const float* __restrict__ g,
                                                      const float* __restrict__ be,
                                                      __bf16* __restrict__ out) {
    __shared__ float red[C];
    int row = blockIdx.x, t = threadIdx.x;
    float v = x[(size_t)row * C + t];
    red[t] = v;
    __syncthreads();
    for (int s = C / 2; s > 0; s >>= 1) { if (t < s) red[t] += red[t + s]; __syncthreads(); }
    float mu = red[0] / (float)C;
    __syncthreads();
    float d = v - mu;
    red[t] = d * d;
    __syncthreads();
    for (int s = C / 2; s > 0; s >>= 1) { if (t < s) red[t] += red[t + s]; __syncthreads(); }
    float var = red[0] / (float)C;
    float y = d * rsqrtf(var + 1e-5f) * g[t] + be[t];
    out[(size_t)row * C + t] = (__bf16)y;
}

// ---------------------------------------------------------------- fused 2-layer MLP (WMMA bf16, f32 accum)
// 64-row tile per block; 8 waves = 4 row-tiles x 2 column groups.
// AMODE 0: A = fp32 srcA_f32[row][DIN]               (input nets)
// AMODE 1: A = bf16 srcA_bf[row][DIN]                (output nets, pre-normalized)
// AMODE 2: A = [nf_bf(256) | agg_bf(128)]            (e2n, DIN=384)
// AMODE 3: A = [ef_bf(128) | nf_bf[i1] | nf_bf[i2]]  (n2e, DIN=640)
template <int DIN, int H, int DOUT, int AMODE, bool RESID, bool MASK>
__global__ __launch_bounds__(256) void mlp2_kernel(
    const float*  __restrict__ srcA_f32,
    const __bf16* __restrict__ srcA_bf,
    const __bf16* __restrict__ srcB_bf,
    const int*    __restrict__ idx1,
    const int*    __restrict__ idx2,
    const __bf16* __restrict__ w1t, const float* __restrict__ b1,
    const __bf16* __restrict__ w2t, const float* __restrict__ b2,
    const float* resid,
    const float* __restrict__ maskp,
    float* outf,
    __bf16* outbf) {
    extern __shared__ char smem[];
    __bf16* ldsA = (__bf16*)smem;                      // 64 x DIN
    __bf16* ldsH = (__bf16*)(smem + 64 * DIN * 2);     // 64 x H
    const int tid  = threadIdx.x;
    const int lane = tid & 31;
    const int wid  = tid >> 5;
    const size_t rowbase = (size_t)blockIdx.x * 64;

    // ------------- stage A tile into LDS (bf16) -------------
    if (AMODE == 0) {
        for (int i = tid; i < 64 * DIN; i += 256)
            ldsA[i] = (__bf16)srcA_f32[(rowbase + i / DIN) * DIN + (i % DIN)];
    } else if (AMODE == 1) {
        for (int i = tid; i < 64 * (DIN / 8); i += 256) {
            int r = i / (DIN / 8), c = (i % (DIN / 8)) * 8;
            *(v8bf*)(ldsA + r * DIN + c) = *(const v8bf*)(srcA_bf + (rowbase + r) * DIN + c);
        }
    } else if (AMODE == 2) {  // DIN = 384 = 48 chunks of 8
        for (int i = tid; i < 64 * 48; i += 256) {
            int r = i / 48, c8 = i % 48;
            size_t grow = rowbase + r;
            v8bf v;
            if (c8 < 32) v = *(const v8bf*)(srcA_bf + grow * 256 + c8 * 8);
            else         v = *(const v8bf*)(srcB_bf + grow * 128 + (c8 - 32) * 8);
            *(v8bf*)(ldsA + r * DIN + c8 * 8) = v;
        }
    } else {                  // AMODE 3, DIN = 640 = 80 chunks of 8
        for (int i = tid; i < 64 * 80; i += 256) {
            int r = i / 80, c8 = i % 80;
            size_t grow = rowbase + r;
            int b = (int)(grow / E_), e = (int)(grow % E_);
            v8bf v;
            if (c8 < 16) {
                v = *(const v8bf*)(srcA_bf + grow * 128 + c8 * 8);
            } else if (c8 < 48) {
                size_t node = (size_t)b * N_ + idx1[e];
                v = *(const v8bf*)(srcB_bf + node * 256 + (c8 - 16) * 8);
            } else {
                size_t node = (size_t)b * N_ + idx2[e];
                v = *(const v8bf*)(srcB_bf + node * 256 + (c8 - 48) * 8);
            }
            *(v8bf*)(ldsA + r * DIN + c8 * 8) = v;
        }
    }
    __syncthreads();

    const int rt   = wid & 3;   // row tile (16 rows)
    const int cg   = wid >> 2;  // column group
    const int row0 = rt * 16;

    // ------------- GEMM1: [64,DIN] x W1t -> bias+GELU -> ldsH bf16 -------------
    constexpr int NJ = H / 32;
    v8f acc[NJ] = {};
    for (int kc = 0; kc < DIN; kc += 32) {
        v16bf a = load_a_frag(ldsA, row0, kc, DIN, lane);
        if (kc + 32 < DIN)
            __builtin_prefetch(w1t + (size_t)(cg * 16 + (lane & 15)) * DIN + kc + 32, 0, 3);
#pragma unroll
        for (int j = 0; j < NJ; j++) {
            int col0 = cg * 16 + j * 32;
            v16bf bf = load_b_frag(w1t, col0, kc, DIN, lane);
            acc[j] = __builtin_amdgcn_wmma_f32_16x16x32_bf16(false, a, false, bf,
                                                             (short)0, acc[j], false, false);
        }
    }
    {
        int l = lane & 15, hr = (lane >> 4) * 8;
#pragma unroll
        for (int j = 0; j < NJ; j++) {
            int col = cg * 16 + j * 32 + l;
            float bv = b1[col];
#pragma unroll
            for (int r = 0; r < 8; r++)
                ldsH[(row0 + hr + r) * H + col] = (__bf16)gelu_f(acc[j][r] + bv);
        }
    }
    __syncthreads();

    // ------------- GEMM2: [64,H] x W2t -> epilogue -> global -------------
    constexpr int NJ2 = DOUT / 32;
    v8f acc2[NJ2] = {};
    for (int kc = 0; kc < H; kc += 32) {
        v16bf a = load_a_frag(ldsH, row0, kc, H, lane);
#pragma unroll
        for (int j = 0; j < NJ2; j++) {
            int col0 = cg * 16 + j * 32;
            v16bf bf = load_b_frag(w2t, col0, kc, H, lane);
            acc2[j] = __builtin_amdgcn_wmma_f32_16x16x32_bf16(false, a, false, bf,
                                                              (short)0, acc2[j], false, false);
        }
    }
    {
        int l = lane & 15, hr = (lane >> 4) * 8;
#pragma unroll
        for (int j = 0; j < NJ2; j++) {
            int col = cg * 16 + j * 32 + l;
            float bv = b2[col];
#pragma unroll
            for (int r = 0; r < 8; r++) {
                size_t grow = rowbase + row0 + hr + r;
                float v = acc2[j][r] + bv;
                if (RESID) v += resid[grow * DOUT + col];
                if (MASK)  v *= maskp[grow];  // mask flat over [B*E]
                outf[grow * DOUT + col] = v;
                if (outbf) outbf[grow * DOUT + col] = (__bf16)v;
            }
        }
    }
}

// ================================================================ host
extern "C" void kernel_launch(void* const* d_in, const int* in_sizes, int n_in,
                              void* d_out, int out_size, void* d_ws, size_t ws_size,
                              hipStream_t stream) {
    (void)in_sizes; (void)n_in; (void)out_size; (void)ws_size;

    const float* z_nodes = (const float*)d_in[0];
    const float* z_edges = (const float*)d_in[1];
    const int*   idx1    = (const int*)d_in[3];
    const int*   idx2    = (const int*)d_in[4];
    const float* mask    = (const float*)d_in[5];

    // params leaves, pytree (sorted-key) order: in_e, in_n, layers[4]{e2n,n2e}, out_e, out_n
    int p = 6;
    const float* ine_b[2]; const float* ine_w[2];
    const float* inn_b[2]; const float* inn_w[2];
    for (int i = 0; i < 2; i++) { ine_b[i] = (const float*)d_in[p++]; ine_w[i] = (const float*)d_in[p++]; }
    for (int i = 0; i < 2; i++) { inn_b[i] = (const float*)d_in[p++]; inn_w[i] = (const float*)d_in[p++]; }
    const float *e2n_b[4][2], *e2n_w[4][2], *n2e_b[4][2], *n2e_w[4][2];
    for (int l = 0; l < 4; l++) {
        for (int i = 0; i < 2; i++) { e2n_b[l][i] = (const float*)d_in[p++]; e2n_w[l][i] = (const float*)d_in[p++]; }
        for (int i = 0; i < 2; i++) { n2e_b[l][i] = (const float*)d_in[p++]; n2e_w[l][i] = (const float*)d_in[p++]; }
    }
    const float* oe_be  = (const float*)d_in[p++];
    const float* oe_g   = (const float*)d_in[p++];
    const float* oe_l1b = (const float*)d_in[p++]; const float* oe_l1w = (const float*)d_in[p++];
    const float* oe_l2b = (const float*)d_in[p++]; const float* oe_l2w = (const float*)d_in[p++];
    const float* on_be  = (const float*)d_in[p++];
    const float* on_g   = (const float*)d_in[p++];
    const float* on_l1b = (const float*)d_in[p++]; const float* on_l1w = (const float*)d_in[p++];
    const float* on_l2b = (const float*)d_in[p++]; const float* on_l2w = (const float*)d_in[p++];

    // ---- workspace carve-out ----
    size_t off = 0;
    auto carve = [&](size_t bytes) -> void* {
        void* ptr = (char*)d_ws + off;
        off += (bytes + 255) & ~(size_t)255;
        return ptr;
    };
    float*  nf     = (float*)carve((size_t)MNODE * HN_ * 4);
    float*  ef     = (float*)carve((size_t)MEDGE * HE_ * 4);
    __bf16* nf_bf  = (__bf16*)carve((size_t)MNODE * HN_ * 2);
    __bf16* ef_bf  = (__bf16*)carve((size_t)MEDGE * HE_ * 2);
    __bf16* agg_bf = (__bf16*)carve((size_t)MNODE * HE_ * 2);

    auto carve_w = [&](int K, int Nc) -> __bf16* { return (__bf16*)carve((size_t)K * Nc * 2); };
    __bf16* t_ine[2]  = { carve_w(CIN_, HE_),  carve_w(HE_, HE_) };
    __bf16* t_inn[2]  = { carve_w(CIN_, HN_),  carve_w(HN_, HN_) };
    __bf16* t_e2n[4][2]; __bf16* t_n2e[4][2];
    for (int l = 0; l < 4; l++) {
        t_e2n[l][0] = carve_w(HN_ + HE_ + HN_ - HE_, HN_);          // 512 x 256 carve (384 x 256 used)
        t_e2n[l][1] = carve_w(HN_, HN_);
        t_n2e[l][0] = carve_w(HE_ + 2 * HN_, HE_);                  // 640 x 128
        t_n2e[l][1] = carve_w(HE_, HE_);
    }
    __bf16* t_oel1 = carve_w(HE_, HE_);
    __bf16* t_oel2 = carve_w(HE_, COUT_);
    __bf16* t_onl1 = carve_w(HN_, HN_);
    __bf16* t_onl2 = carve_w(HN_, COUT_);

    // ---- weight conversion (fp32 [K][N] -> bf16 [N][K]) ----
    struct WD { const float* s; __bf16* d; int K; int Nc; };
    WD wds[24] = {
        { ine_w[0], t_ine[0], CIN_, HE_ }, { ine_w[1], t_ine[1], HE_, HE_ },
        { inn_w[0], t_inn[0], CIN_, HN_ }, { inn_w[1], t_inn[1], HN_, HN_ },
        { e2n_w[0][0], t_e2n[0][0], 384, HN_ }, { e2n_w[0][1], t_e2n[0][1], HN_, HN_ },
        { n2e_w[0][0], t_n2e[0][0], 640, HE_ }, { n2e_w[0][1], t_n2e[0][1], HE_, HE_ },
        { e2n_w[1][0], t_e2n[1][0], 384, HN_ }, { e2n_w[1][1], t_e2n[1][1], HN_, HN_ },
        { n2e_w[1][0], t_n2e[1][0], 640, HE_ }, { n2e_w[1][1], t_n2e[1][1], HE_, HE_ },
        { e2n_w[2][0], t_e2n[2][0], 384, HN_ }, { e2n_w[2][1], t_e2n[2][1], HN_, HN_ },
        { n2e_w[2][0], t_n2e[2][0], 640, HE_ }, { n2e_w[2][1], t_n2e[2][1], HE_, HE_ },
        { e2n_w[3][0], t_e2n[3][0], 384, HN_ }, { e2n_w[3][1], t_e2n[3][1], HN_, HN_ },
        { n2e_w[3][0], t_n2e[3][0], 640, HE_ }, { n2e_w[3][1], t_n2e[3][1], HE_, HE_ },
        { oe_l1w, t_oel1, HE_, HE_ }, { oe_l2w, t_oel2, HE_, COUT_ },
        { on_l1w, t_onl1, HN_, HN_ }, { on_l2w, t_onl2, HN_, COUT_ },
    };
    for (int i = 0; i < 24; i++) {
        int total = wds[i].K * wds[i].Nc;
        wprep_kernel<<<(total + 255) / 256, 256, 0, stream>>>(wds[i].s, wds[i].d, wds[i].K, wds[i].Nc);
    }

    // ---- input MLPs ----
    mlp2_kernel<64, 256, 256, 0, false, false>
        <<<MNODE / 64, 256, (64 * 64 + 64 * 256) * 2, stream>>>(
            z_nodes, nullptr, nullptr, nullptr, nullptr,
            t_inn[0], inn_b[0], t_inn[1], inn_b[1],
            nullptr, nullptr, nf, nf_bf);
    mlp2_kernel<64, 128, 128, 0, false, false>
        <<<MEDGE / 64, 256, (64 * 64 + 64 * 128) * 2, stream>>>(
            z_edges, nullptr, nullptr, nullptr, nullptr,
            t_ine[0], ine_b[0], t_ine[1], ine_b[1],
            nullptr, nullptr, ef, ef_bf);

    // ---- message-passing layers ----
    for (int l = 0; l < 4; l++) {
        agg_kernel<<<MNODE, 128, 0, stream>>>(ef, mask, agg_bf);
        mlp2_kernel<384, 256, 256, 2, true, false>
            <<<MNODE / 64, 256, (64 * 384 + 64 * 256) * 2, stream>>>(
                nullptr, nf_bf, agg_bf, nullptr, nullptr,
                t_e2n[l][0], e2n_b[l][0], t_e2n[l][1], e2n_b[l][1],
                nf, nullptr, nf, nf_bf);
        mlp2_kernel<640, 128, 128, 3, true, true>
            <<<MEDGE / 64, 256, (64 * 640 + 64 * 128) * 2, stream>>>(
                nullptr, ef_bf, nf_bf, idx1, idx2,
                t_n2e[l][0], n2e_b[l][0], t_n2e[l][1], n2e_b[l][1],
                ef, mask, ef, ef_bf);
    }

    // ---- output nets ----
    float* out_nodes = (float*)d_out;
    float* out_edges = out_nodes + (size_t)MNODE * COUT_;

    layernorm_kernel<256><<<MNODE, 256, 0, stream>>>(nf, on_g, on_be, nf_bf);
    mlp2_kernel<256, 256, 64, 1, false, false>
        <<<MNODE / 64, 256, (64 * 256 + 64 * 256) * 2, stream>>>(
            nullptr, nf_bf, nullptr, nullptr, nullptr,
            t_onl1, on_l1b, t_onl2, on_l2b,
            nullptr, nullptr, out_nodes, nullptr);

    layernorm_kernel<128><<<MEDGE, 128, 0, stream>>>(ef, oe_g, oe_be, ef_bf);
    mlp2_kernel<128, 128, 64, 1, false, true>
        <<<MEDGE / 64, 256, (64 * 128 + 64 * 128) * 2, stream>>>(
            nullptr, ef_bf, nullptr, nullptr, nullptr,
            t_oel1, oe_l1b, t_oel2, oe_l2b,
            nullptr, mask, out_edges, nullptr);
}
